// QuantumLayer_82377472737503
// MI455X (gfx1250) — compile-verified, matrix operand order checked by reading, not run on hardware
//
#include <hip/hip_runtime.h>
#include <hip/hip_bf16.h>

// CDNA5 (gfx1250) wave32 WMMA types
typedef __attribute__((ext_vector_type(16))) _Float16 v16h;
typedef __attribute__((ext_vector_type(8)))  float    v8f;

#define N_QUBITS 8
#define ROWS_PER_WAVE 16   // one 16x16x32 WMMA tile of batch rows per wave
#define WAVES_PER_BLOCK 8  // 256 threads
#define ROWS_PER_BLOCK (ROWS_PER_WAVE * WAVES_PER_BLOCK)

// out[b, w] = |prod of diagonal phases|^2 at basis index 0, replicated over the
// 8 wires (Z_DIAG row 0 is all +1). theta_b = -0.5*(sum_k x[b,k] + sum(all weights)).
// The -0.5 * row-sum is computed with v_wmma_f32_16x16x32_f16 (B = -0.5 splat).
__global__ __launch_bounds__(256) void quantum_z_kernel(
    const float* __restrict__ x,   // [B, 8]
    const float* __restrict__ w,   // [2, 8] -> 16 floats
    float* __restrict__ out,       // [B, 8]
    int B)
{
    const int lane = threadIdx.x & 31;
    const int wave = threadIdx.x >> 5;
    const int rowBase = ((int)blockIdx.x * WAVES_PER_BLOCK + wave) * ROWS_PER_WAVE;
    if (rowBase >= B) return;  // wave-uniform exit: surviving waves keep EXEC all-1s

    // Uniform weight sum (scalar loads; weights are tiny and L2-resident).
    float wsum = 0.0f;
#pragma unroll
    for (int i = 0; i < 16; ++i) wsum += w[i];

    // A matrix: 16-bit A 16x32 layout. Lanes 0-15 hold row m=lane, K=0..7 in
    // VGPR halves 0..7 (v16h elements 0..7). K=8..31 zero (lanes 16-31 all zero).
    v16h a;
#pragma unroll
    for (int i = 0; i < 16; ++i) a[i] = (_Float16)0.0f;

    if (lane < 16) {
        const int row = rowBase + lane;
        if (row < B) {
            const float4* px = (const float4*)(x + (size_t)row * N_QUBITS);
            float4 x0 = px[0];
            float4 x1 = px[1];
            a[0] = (_Float16)x0.x; a[1] = (_Float16)x0.y;
            a[2] = (_Float16)x0.z; a[3] = (_Float16)x0.w;
            a[4] = (_Float16)x1.x; a[5] = (_Float16)x1.y;
            a[6] = (_Float16)x1.z; a[7] = (_Float16)x1.w;
        }
    }

    // B matrix: constant -0.5 everywhere -> layout-independent.
    v16h bm;
#pragma unroll
    for (int i = 0; i < 16; ++i) bm[i] = (_Float16)(-0.5f);

    v8f c = {};
    // D[m][n] = sum_k A[m][k] * (-0.5) = -0.5 * sum_k x[rowBase+m][k]
    c = __builtin_amdgcn_wmma_f32_16x16x32_f16(
        /*neg_a=*/false, a, /*neg_b=*/false, bm,
        /*c_mod=*/(short)0, c, /*reuse_a=*/false, /*reuse_b=*/false);

    // C layout: VGPR r = row r (lanes 0-15) / row 8+r (lanes 16-31), replicated
    // across the 16 lanes of each half. Compute p_r = cos^2 + sin^2 per slot.
    float pr[8];
#pragma unroll
    for (int r = 0; r < 8; ++r) {
        float th = c[r] - 0.5f * wsum;
        float cc = __cosf(th);
        float ss = __sinf(th);
        pr[r] = cc * cc + ss * ss;
    }

    // Lane L stores float4 covering row rowBase + L/2, cols (L&1)*4..+3.
    // Needed value p_{L/2} is in this lane's pr[(L>>1)&7] (both lane halves).
    const int sel = (lane >> 1) & 7;
    float s0 = (sel & 1) ? pr[1] : pr[0];
    float s1 = (sel & 1) ? pr[3] : pr[2];
    float s2 = (sel & 1) ? pr[5] : pr[4];
    float s3 = (sel & 1) ? pr[7] : pr[6];
    float t0 = (sel & 2) ? s1 : s0;
    float t1 = (sel & 2) ? s3 : s2;
    float p  = (sel & 4) ? t1 : t0;

    const int rowOfLane = rowBase + (lane >> 1);
    if (rowOfLane < B) {
        float4 o; o.x = p; o.y = p; o.z = p; o.w = p;
        // wave writes one contiguous 512B block: rows rowBase..rowBase+15
        ((float4*)out)[(size_t)rowBase * 2 + lane] = o;
    }
}

extern "C" void kernel_launch(void* const* d_in, const int* in_sizes, int n_in,
                              void* d_out, int out_size, void* d_ws, size_t ws_size,
                              hipStream_t stream) {
    const float* x = (const float*)d_in[0];   // (64,1024,8) f32
    const float* w = (const float*)d_in[1];   // (2,8) f32
    float* out = (float*)d_out;               // (64,1024,8) f32

    const int B = in_sizes[0] / N_QUBITS;     // 65536 rows
    const int blocks = (B + ROWS_PER_BLOCK - 1) / ROWS_PER_BLOCK;
    quantum_z_kernel<<<blocks, WAVES_PER_BLOCK * 32, 0, stream>>>(x, w, out, B);
}
